// DeformableTransformer_21638045237363
// MI455X (gfx1250) — compile-verified, matrix operand order checked by reading, not run on hardware
//
#include <hip/hip_runtime.h>
#include <hip/hip_bf16.h>

// ---------------------------------------------------------------------------
// Problem constants (match the reference)
// ---------------------------------------------------------------------------
#define BB    4
#define QQ    800
#define DD    256
#define NHH   8
#define DHH   32
#define NLL   4
#define NPP   4
#define NLAYERS 6
#define DFFF  1024
#define SS    21760          // 128*128 + 64*64 + 32*32 + 16*16
#define BQ    (BB*QQ)        // 3200
#define BS    (BB*SS)        // 87040

typedef __attribute__((ext_vector_type(16))) _Float16 v16h;
typedef __attribute__((ext_vector_type(8)))  float    v8f;
typedef __attribute__((ext_vector_type(4)))  int      i32x4;
typedef __attribute__((ext_vector_type(8)))  int      i32x8;

// ---------------------------------------------------------------------------
// WMMA helpers (wave32, D = A(16x32,f16) * B(32x16,f16) + C(16x16,f32))
// A-frag: lane&15 = row, element e -> k = (e<8?e:e+8) + 8*(lane>>4)
// B-frag: lane&15 = col, same k pattern (W is row-major N x K, row = col n)
// C:      lane&15 = col, vgpr r -> row = r + 8*(lane>>4)
// ---------------------------------------------------------------------------
__device__ __forceinline__ v8f wmma_f16(v16h a, v16h b, v8f c) {
  return __builtin_amdgcn_wmma_f32_16x16x32_f16(false, a, false, b,
                                                (short)0, c, false, false);
}

// Build a fragment from a row-major f32 row pointer (global). koff = 8*half.
__device__ __forceinline__ v16h load_frag_f32(const float* __restrict__ rowp,
                                              int koff) {
  const float4* p = reinterpret_cast<const float4*>(rowp + koff);
  const float4* q = reinterpret_cast<const float4*>(rowp + koff + 16);
  float4 a0 = p[0], a1 = p[1], b0 = q[0], b1 = q[1];
  v16h f;
  f[0]=(_Float16)a0.x; f[1]=(_Float16)a0.y; f[2]=(_Float16)a0.z; f[3]=(_Float16)a0.w;
  f[4]=(_Float16)a1.x; f[5]=(_Float16)a1.y; f[6]=(_Float16)a1.z; f[7]=(_Float16)a1.w;
  f[8]=(_Float16)b0.x; f[9]=(_Float16)b0.y; f[10]=(_Float16)b0.z; f[11]=(_Float16)b0.w;
  f[12]=(_Float16)b1.x; f[13]=(_Float16)b1.y; f[14]=(_Float16)b1.z; f[15]=(_Float16)b1.w;
  return f;
}

// Fragment from an LDS row (f32, padded row stride). 16B ds_load_b128 chunks.
__device__ __forceinline__ v16h frag_from_lds(const float* __restrict__ rp,
                                              int half) {
  const float4 a = *reinterpret_cast<const float4*>(rp + 8 * half);
  const float4 b = *reinterpret_cast<const float4*>(rp + 8 * half + 4);
  const float4 c = *reinterpret_cast<const float4*>(rp + 8 * half + 16);
  const float4 d = *reinterpret_cast<const float4*>(rp + 8 * half + 20);
  v16h f;
  f[0]=(_Float16)a.x; f[1]=(_Float16)a.y; f[2]=(_Float16)a.z; f[3]=(_Float16)a.w;
  f[4]=(_Float16)b.x; f[5]=(_Float16)b.y; f[6]=(_Float16)b.z; f[7]=(_Float16)b.w;
  f[8]=(_Float16)c.x; f[9]=(_Float16)c.y; f[10]=(_Float16)c.z; f[11]=(_Float16)c.w;
  f[12]=(_Float16)d.x; f[13]=(_Float16)d.y; f[14]=(_Float16)d.z; f[15]=(_Float16)d.w;
  return f;
}

// ---------------------------------------------------------------------------
// Tensor Data Mover: 2-D tile load (rows x 32 f32) into LDS with padding
// (+4 DWORDs every 32 DWORDs -> LDS row stride 36, bank-conflict-free).
// Descriptor per CDNA5 ISA; tracked by TENSORcnt.
// ---------------------------------------------------------------------------
#define LDS_STRIDE 36

__device__ __forceinline__ unsigned lds_off_u32(const void* p) {
  return (unsigned)(uintptr_t)(__attribute__((address_space(3))) const void*)p;
}

__device__ __forceinline__ void tdm_load_tile(const float* gsrc,
                                              unsigned ldsByte,
                                              int tileRows, int Kdim,
                                              int totRows) {
  const unsigned long long ga = (unsigned long long)(uintptr_t)gsrc;
  i32x4 g0;
  g0[0] = 1;                                   // count=1 (valid descriptor)
  g0[1] = (int)ldsByte;                        // lds_addr (bytes)
  g0[2] = (int)(unsigned)(ga & 0xffffffffull); // global_addr[31:0]
  g0[3] = (int)(((unsigned)(ga >> 32) & 0x01ffffffu) | 0x80000000u); // hi|type=2
  i32x8 g1;
  // data_size=4B (2<<16) | pad_enable (1<<20) | pad_interval=32dw (4<<22)
  // | pad_amount=4dw (3<<25)
  g1[0] = (int)(0x00020000u | 0x00100000u | (4u << 22) | (3u << 25));
  g1[1] = (int)(((unsigned)Kdim & 0xffffu) << 16);                 // dim0 lo16
  g1[2] = (int)((((unsigned)Kdim >> 16) & 0xffffu) |
                (((unsigned)totRows & 0xffffu) << 16));            // dim0 hi|dim1 lo
  g1[3] = (int)((((unsigned)totRows >> 16) & 0xffffu) | (32u << 16)); // tile_dim0=32
  g1[4] = (int)((unsigned)tileRows & 0xffffu);                     // tile_dim1
  g1[5] = (int)(unsigned)Kdim;                                     // dim0_stride lo32
  g1[6] = 0;
  g1[7] = 0;
  asm volatile("tensor_load_to_lds %0, %1" :: "s"(g0), "s"(g1) : "memory");
}

// ---------------------------------------------------------------------------
// GEMM: C[M,N] = A[M,K] * W[N,K]^T + bias (+ optional ReLU)
// Requires M%128==0, N%64==0, K%32==0 (true at every call site).
// Block tile 128x64, 8 waves as 4(M)x2(N), wave tile 32x32 (2x2 WMMA tiles).
// A/W slabs TDM-staged into LDS, double-buffered.
// ---------------------------------------------------------------------------
__global__ __launch_bounds__(256) void gemm_wmma_kernel(
    const float* __restrict__ A, const float* __restrict__ W,
    const float* __restrict__ bias, float* __restrict__ C,
    int M, int N, int K, int relu)
{
  __shared__ float ldsA[2][128 * LDS_STRIDE];   // 2 x 18 KB
  __shared__ float ldsW[2][64 * LDS_STRIDE];    // 2 x  9 KB

  const int tid = threadIdx.x;
  const int wave = tid >> 5, lane = tid & 31;
  const int half = lane >> 4, l = lane & 15;
  const int wm = wave & 3, wn = wave >> 2;
  const int mBase = blockIdx.y * 128;
  const int nBase = blockIdx.x * 64;

  if (wave == 0)
    tdm_load_tile(A + (size_t)mBase * K, lds_off_u32(&ldsA[0][0]), 128, K, M);
  if (wave == 1)
    tdm_load_tile(W + (size_t)nBase * K, lds_off_u32(&ldsW[0][0]), 64, K, N);

  v8f acc[2][2] = {};
  int buf = 0;
  for (int k0 = 0; k0 < K; k0 += 32) {
    if (k0 + 32 < K) {      // prefetch next K-slab into the other buffer
      if (wave == 0)
        tdm_load_tile(A + (size_t)mBase * K + (k0 + 32),
                      lds_off_u32(&ldsA[buf ^ 1][0]), 128, K, M);
      if (wave == 1)
        tdm_load_tile(W + (size_t)nBase * K + (k0 + 32),
                      lds_off_u32(&ldsW[buf ^ 1][0]), 64, K, N);
      if (wave < 2) __builtin_amdgcn_s_wait_tensorcnt(1);
    } else {
      if (wave < 2) __builtin_amdgcn_s_wait_tensorcnt(0);
    }
    __syncthreads();        // current buffer visible to all waves

    const float* at = &ldsA[buf][0];
    const float* wt = &ldsW[buf][0];
    v16h afrag[2], bfrag[2];
#pragma unroll
    for (int t = 0; t < 2; ++t) {
      afrag[t] = frag_from_lds(at + (wm * 32 + t * 16 + l) * LDS_STRIDE, half);
      bfrag[t] = frag_from_lds(wt + (wn * 32 + t * 16 + l) * LDS_STRIDE, half);
    }
#pragma unroll
    for (int t = 0; t < 2; ++t)
#pragma unroll
      for (int u = 0; u < 2; ++u)
        acc[t][u] = wmma_f16(afrag[t], bfrag[u], acc[t][u]);

    __syncthreads();        // all reads done before TDM overwrites buffer
    buf ^= 1;
  }

#pragma unroll
  for (int t = 0; t < 2; ++t) {
    const int mt = mBase + wm * 32 + t * 16;
#pragma unroll
    for (int u = 0; u < 2; ++u) {
      const int col = nBase + wn * 32 + u * 16 + l;
      const float bv = bias ? bias[col] : 0.0f;
#pragma unroll
      for (int r = 0; r < 8; ++r) {
        const int row = mt + r + 8 * half;
        float v = acc[t][u][r] + bv;
        if (relu) v = fmaxf(v, 0.0f);
        C[(size_t)row * N + col] = v;
      }
    }
  }
}

// ---------------------------------------------------------------------------
// Self-attention: one workgroup per (b, h, 16-query tile).
// scores (16x800) via WMMA -> LDS, softmax in LDS, P*V via WMMA with
// ds_add_f32 combination of per-wave partials.
// ---------------------------------------------------------------------------
__global__ __launch_bounds__(256) void attn_kernel(
    const float* __restrict__ QKV,   // (B*Q, 768) [q|k|v]
    float* __restrict__ O)           // (B*Q, 256)
{
  __shared__ float sc[16 * QQ];      // 51200 B
  __shared__ float ot[16 * 32];      // 2048 B

  const int b = blockIdx.z, h = blockIdx.y, q0 = blockIdx.x * 16;
  const int tid = threadIdx.x;
  const int wave = tid >> 5, lane = tid & 31;
  const int half = lane >> 4, l = lane & 15;

  ot[tid]       = 0.0f;
  ot[tid + 256] = 0.0f;

  // ---- scores = (Q . K^T) / sqrt(32): 50 key tiles of 16 split over 8 waves
  const float* qrow = QKV + (size_t)(b * QQ + q0 + l) * 768 + h * DHH;
  const v16h aq = load_frag_f32(qrow, 8 * half);   // one K-step (DH == 32)
  const float scale = 0.1767766952966369f;         // 1/sqrt(32)

  for (int kb = wave; kb < QQ / 16; kb += 8) {
    const float* krow = QKV + (size_t)(b * QQ + kb * 16 + l) * 768 + DD + h * DHH;
    const v16h bk = load_frag_f32(krow, 8 * half);
    v8f c = {};
    c = wmma_f16(aq, bk, c);
#pragma unroll
    for (int r = 0; r < 8; ++r) {
      const int m = r + 8 * half;
      sc[m * QQ + kb * 16 + l] = c[r] * scale;
    }
  }
  __syncthreads();

  // ---- softmax over 800 keys: 16 threads per row, shuffle-reduce inside row
  {
    const int row = tid >> 4, j = tid & 15;
    float mx = -1e30f;
    for (int i = j; i < QQ; i += 16) mx = fmaxf(mx, sc[row * QQ + i]);
    mx = fmaxf(mx, __shfl_xor(mx, 8, 32));
    mx = fmaxf(mx, __shfl_xor(mx, 4, 32));
    mx = fmaxf(mx, __shfl_xor(mx, 2, 32));
    mx = fmaxf(mx, __shfl_xor(mx, 1, 32));
    float sum = 0.0f;
    for (int i = j; i < QQ; i += 16) {
      const float e = __expf(sc[row * QQ + i] - mx);
      sc[row * QQ + i] = e;
      sum += e;
    }
    sum += __shfl_xor(sum, 8, 32);
    sum += __shfl_xor(sum, 4, 32);
    sum += __shfl_xor(sum, 2, 32);
    sum += __shfl_xor(sum, 1, 32);
    const float inv = 1.0f / sum;
    for (int i = j; i < QQ; i += 16) sc[row * QQ + i] *= inv;
  }
  __syncthreads();

  // ---- O(16x32) = P(16x800) * V(800x32); 25 key blocks of 32 over 8 waves
  v8f c0 = {}, c1 = {};
  for (int kb = wave; kb < QQ / 32; kb += 8) {
    v16h ap, bv0, bv1;
#pragma unroll
    for (int e = 0; e < 16; ++e) {
      const int k = (e < 8 ? e : e + 8) + 8 * half;
      ap[e] = (_Float16)sc[l * QQ + kb * 32 + k];
      const float* vr = QKV + (size_t)(b * QQ + kb * 32 + k) * 768 + 2 * DD + h * DHH;
      bv0[e] = (_Float16)vr[l];
      bv1[e] = (_Float16)vr[16 + l];
    }
    c0 = wmma_f16(ap, bv0, c0);
    c1 = wmma_f16(ap, bv1, c1);
  }
#pragma unroll
  for (int r = 0; r < 8; ++r) {
    const int m = r + 8 * half;
    atomicAdd(&ot[m * 32 + l], c0[r]);        // ds_add_f32
    atomicAdd(&ot[m * 32 + 16 + l], c1[r]);
  }
  __syncthreads();

  for (int i = tid; i < 512; i += 256) {
    const int m = i >> 5, d = i & 31;
    O[(size_t)(b * QQ + q0 + m) * DD + h * DHH + d] = ot[i];
  }
}

// ---------------------------------------------------------------------------
// x = LayerNorm(x + y) * g + b  (one token per wave, D=256 -> 8 per lane)
// ---------------------------------------------------------------------------
__global__ __launch_bounds__(256) void ln_residual_kernel(
    float* __restrict__ x, const float* __restrict__ y,
    const float* __restrict__ g, const float* __restrict__ beta, int ntok)
{
  const int wave = threadIdx.x >> 5, lane = threadIdx.x & 31;
  const int tok = blockIdx.x * 8 + wave;
  if (tok >= ntok) return;
  const float* xr = x + (size_t)tok * DD;
  const float* yr = y + (size_t)tok * DD;
  float r[8];
  float s = 0.0f;
#pragma unroll
  for (int i = 0; i < 8; ++i) {
    r[i] = xr[lane + 32 * i] + yr[lane + 32 * i];
    s += r[i];
  }
#pragma unroll
  for (int m = 16; m >= 1; m >>= 1) s += __shfl_xor(s, m, 32);
  const float mu = s * (1.0f / 256.0f);
  float v = 0.0f;
#pragma unroll
  for (int i = 0; i < 8; ++i) { const float d = r[i] - mu; v += d * d; }
#pragma unroll
  for (int m = 16; m >= 1; m >>= 1) v += __shfl_xor(v, m, 32);
  const float rstd = rsqrtf(v * (1.0f / 256.0f) + 1e-5f);
#pragma unroll
  for (int i = 0; i < 8; ++i) {
    const int d = lane + 32 * i;
    x[(size_t)tok * DD + d] = (r[i] - mu) * rstd * g[d] + beta[d];
  }
}

// ---------------------------------------------------------------------------
// Sampling prep: per (b,q,h) -> softmax(aw, 16) and sampling locations
// ---------------------------------------------------------------------------
__global__ __launch_bounds__(256) void prep_sampling_kernel(
    const float* __restrict__ offr,   // (B*Q, 256)
    const float* __restrict__ awr,    // (B*Q, 128)
    const float* __restrict__ ref,    // (B*Q, 2)
    const float* __restrict__ valid,  // (B, NL, 2)
    float* __restrict__ loc,          // (B*Q, NH, 16, 2)
    float* __restrict__ aw)           // (B*Q, NH, 16)
{
  const int idx = blockIdx.x * 256 + threadIdx.x;   // (b*Q + q)*NH + h
  if (idx >= BQ * NHH) return;
  const int h = idx & (NHH - 1);
  const int bq = idx >> 3;
  const int b = bq / QQ;

  const float* offp = offr + (size_t)bq * 256 + h * 32;
  const float* awp  = awr + (size_t)bq * 128 + h * 16;
  const float rx = ref[bq * 2], ry = ref[bq * 2 + 1];

  float mx = -1e30f;
#pragma unroll
  for (int i = 0; i < 16; ++i) mx = fmaxf(mx, awp[i]);
  float e[16], sum = 0.0f;
#pragma unroll
  for (int i = 0; i < 16; ++i) { e[i] = __expf(awp[i] - mx); sum += e[i]; }
  const float inv = 1.0f / sum;

  const int HH[4] = {128, 64, 32, 16};
  const int WW[4] = {128, 64, 32, 16};
#pragma unroll
  for (int lvl = 0; lvl < NLL; ++lvl) {
    const float vrx = valid[(b * NLL + lvl) * 2];
    const float vry = valid[(b * NLL + lvl) * 2 + 1];
    const float bx = rx * vrx, by = ry * vry;
    const float iw = 1.0f / (float)WW[lvl], ih = 1.0f / (float)HH[lvl];
#pragma unroll
    for (int p = 0; p < NPP; ++p) {
      const int o = lvl * NPP + p;
      const size_t w = (size_t)idx * 16 + o;
      loc[w * 2]     = bx + offp[o * 2]     * iw;
      loc[w * 2 + 1] = by + offp[o * 2 + 1] * ih;
      aw[w] = e[o] * inv;
    }
  }
}

// ---------------------------------------------------------------------------
// Deformable sampling: block per (b,q), thread = (head, channel)
// ---------------------------------------------------------------------------
__global__ __launch_bounds__(256) void deform_sample_kernel(
    const float* __restrict__ value,  // (B, S, NH, DH) == (B*S, 256)
    const float* __restrict__ loc,    // (B*Q, NH, 16, 2)
    const float* __restrict__ aw,     // (B*Q, NH, 16)
    float* __restrict__ out)          // (B*Q, 256)
{
  const int bq = blockIdx.x;
  const int b = bq / QQ;
  const int t = threadIdx.x;
  const int h = t >> 5, d = t & 31;

  const int HH[4] = {128, 64, 32, 16};
  const int WW[4] = {128, 64, 32, 16};
  const int ST[4] = {0, 16384, 20480, 21504};

  const size_t base = ((size_t)bq * NHH + h) * 16;
  const float* vb = value + (size_t)b * SS * DD + h * DHH + d;

  float acc = 0.0f;
#pragma unroll
  for (int lvl = 0; lvl < NLL; ++lvl) {
    const int H_ = HH[lvl], W_ = WW[lvl], st = ST[lvl];
#pragma unroll
    for (int p = 0; p < NPP; ++p) {
      const int o = lvl * NPP + p;
      const float lx = loc[(base + o) * 2];
      const float ly = loc[(base + o) * 2 + 1];
      const float wgt = aw[base + o];
      const float x = lx * (float)W_ - 0.5f;
      const float y = ly * (float)H_ - 0.5f;
      const float x0 = floorf(x), y0 = floorf(y);
      const float fx = x - x0, fy = y - y0;
      const int xi = (int)x0, yi = (int)y0;
      float s = 0.0f;
#pragma unroll
      for (int dy = 0; dy < 2; ++dy) {
#pragma unroll
        for (int dx = 0; dx < 2; ++dx) {
          const int xx = xi + dx, yy = yi + dy;
          if (xx >= 0 && xx < W_ && yy >= 0 && yy < H_) {
            const float wb = (dx ? fx : 1.0f - fx) * (dy ? fy : 1.0f - fy);
            s += wb * vb[(size_t)(st + yy * W_ + xx) * DD];
          }
        }
      }
      acc += wgt * s;
    }
  }
  out[(size_t)bq * DD + t] = acc;
}

// ---------------------------------------------------------------------------
// ref <- sigmoid(x @ coord_w^T + coord_b + inverse_sigmoid(ref))
// ---------------------------------------------------------------------------
__global__ __launch_bounds__(256) void coord_update_kernel(
    const float* __restrict__ x, const float* __restrict__ cw,
    const float* __restrict__ cb, float* __restrict__ ref)
{
  const int bq = blockIdx.x * 256 + threadIdx.x;
  if (bq >= BQ) return;
  const float* xr = x + (size_t)bq * DD;
  float o0 = cb[0], o1 = cb[1];
  for (int i = 0; i < DD; ++i) {
    const float xv = xr[i];
    o0 += xv * cw[i];
    o1 += xv * cw[DD + i];
  }
#pragma unroll
  for (int c = 0; c < 2; ++c) {
    float r = ref[bq * 2 + c];
    r = fminf(fmaxf(r, 0.0f), 1.0f);
    const float num = fminf(fmaxf(r, 1e-5f), 1.0f);
    const float den = fminf(fmaxf(1.0f - r, 1e-5f), 1.0f);
    const float inv = logf(num / den);
    const float z = (c == 0 ? o0 : o1) + inv;
    ref[bq * 2 + c] = 1.0f / (1.0f + __expf(-z));
  }
}

// ---------------------------------------------------------------------------
// Finalize: write (output, ref, classes) into d_out (block per token)
// ---------------------------------------------------------------------------
__global__ __launch_bounds__(256) void finalize_kernel(
    const float* __restrict__ x, const float* __restrict__ ref,
    const float* __restrict__ clsw, const float* __restrict__ clsb,
    float* __restrict__ out)
{
  __shared__ float red[8];
  const int bq = blockIdx.x, t = threadIdx.x;
  const float xv = x[(size_t)bq * DD + t];
  out[(size_t)bq * DD + t] = xv;
  float p = xv * clsw[t];
#pragma unroll
  for (int m = 16; m >= 1; m >>= 1) p += __shfl_xor(p, m, 32);
  if ((t & 31) == 0) red[t >> 5] = p;
  __syncthreads();
  if (t == 0) {
    float s = clsb[0];
#pragma unroll
    for (int i = 0; i < 8; ++i) s += red[i];
    out[(size_t)BQ * DD + BQ * 2 + bq] = s;           // point_classes
  }
  if (t < 2) out[(size_t)BQ * DD + bq * 2 + t] = ref[bq * 2 + t];  // ref
}

// ---------------------------------------------------------------------------
// Host-side orchestration
// ---------------------------------------------------------------------------
extern "C" void kernel_launch(void* const* d_in, const int* in_sizes, int n_in,
                              void* d_out, int out_size, void* d_ws, size_t ws_size,
                              hipStream_t stream) {
  (void)in_sizes; (void)n_in; (void)out_size; (void)ws_size;

  // Inputs: top-level dict in insertion order; params pytree leaves sorted by key.
  const float* tgt   = (const float*)d_in[0];
  const float* refin = (const float*)d_in[1];
  const float* src   = (const float*)d_in[2];
  const float* valid = (const float*)d_in[3];
  const float* attn_out_b = (const float*)d_in[4];
  const float* attn_out_w = (const float*)d_in[5];
  const float* aw_b  = (const float*)d_in[6];
  const float* aw_w  = (const float*)d_in[7];
  const float* cls_b = (const float*)d_in[8];
  const float* cls_w = (const float*)d_in[9];
  const float* coord_b = (const float*)d_in[10];
  const float* coord_w = (const float*)d_in[11];
  const float* ffn1_b = (const float*)d_in[12];
  const float* ffn1_w = (const float*)d_in[13];
  const float* ffn2_b = (const float*)d_in[14];
  const float* ffn2_w = (const float*)d_in[15];
  const float* n1_b = (const float*)d_in[16];
  const float* n1_g = (const float*)d_in[17];
  const float* n2_b = (const float*)d_in[18];
  const float* n2_g = (const float*)d_in[19];
  const float* n3_b = (const float*)d_in[20];
  const float* n3_g = (const float*)d_in[21];
  const float* off_b = (const float*)d_in[22];
  const float* off_w = (const float*)d_in[23];
  const float* out_b = (const float*)d_in[24];
  const float* out_w = (const float*)d_in[25];
  const float* qkv_b = (const float*)d_in[26];
  const float* qkv_w = (const float*)d_in[27];
  const float* val_b = (const float*)d_in[28];
  const float* val_w = (const float*)d_in[29];

  float* ws = (float*)d_ws;
  // Workspace layout (floats)
  float* X    = ws;                         // (B*Q, 256)
  float* T2   = X    + (size_t)BQ * DD;     // (B*Q, 256)
  float* QKV  = T2   + (size_t)BQ * DD;     // (B*Q, 768)
  float* ATTO = QKV  + (size_t)BQ * 3 * DD; // (B*Q, 256)
  float* OFFR = ATTO + (size_t)BQ * DD;     // (B*Q, 256)
  float* AWR  = OFFR + (size_t)BQ * DD;     // (B*Q, 128)
  float* LOC  = AWR  + (size_t)BQ * 128;    // (B*Q, NH, 16, 2)
  float* AWN  = LOC  + (size_t)BQ * 256;    // (B*Q, NH, 16)
  float* DEF  = AWN  + (size_t)BQ * 128;    // (B*Q, 256)
  float* HID  = DEF  + (size_t)BQ * DD;     // (B*Q, 1024)
  float* REF  = HID  + (size_t)BQ * DFFF;   // (B*Q, 2)
  float* VAL  = REF  + (size_t)BQ * 2;      // (B*S, 256)

  hipMemcpyAsync(X,   tgt,   (size_t)BQ * DD * sizeof(float), hipMemcpyDeviceToDevice, stream);
  hipMemcpyAsync(REF, refin, (size_t)BQ * 2 * sizeof(float),  hipMemcpyDeviceToDevice, stream);

  const dim3 blk(256);
  for (int lid = 0; lid < NLAYERS; ++lid) {
    const float* l_qkv_w = qkv_w + (size_t)lid * 3 * DD * DD;
    const float* l_qkv_b = qkv_b + (size_t)lid * 3 * DD;
    const float* l_ao_w  = attn_out_w + (size_t)lid * DD * DD;
    const float* l_ao_b  = attn_out_b + (size_t)lid * DD;
    const float* l_off_w = off_w + (size_t)lid * 256 * DD;
    const float* l_off_b = off_b + (size_t)lid * 256;
    const float* l_aw_w  = aw_w + (size_t)lid * 128 * DD;
    const float* l_aw_b  = aw_b + (size_t)lid * 128;
    const float* l_val_w = val_w + (size_t)lid * DD * DD;
    const float* l_val_b = val_b + (size_t)lid * DD;
    const float* l_out_w = out_w + (size_t)lid * DD * DD;
    const float* l_out_b = out_b + (size_t)lid * DD;
    const float* l_f1_w  = ffn1_w + (size_t)lid * DFFF * DD;
    const float* l_f1_b  = ffn1_b + (size_t)lid * DFFF;
    const float* l_f2_w  = ffn2_w + (size_t)lid * DD * DFFF;
    const float* l_f2_b  = ffn2_b + (size_t)lid * DD;
    const float* l_cw    = coord_w + (size_t)lid * 2 * DD;
    const float* l_cb    = coord_b + (size_t)lid * 2;

    // ---- self-attention: QKV proj -> attention -> out proj -> add&LN(n2)
    gemm_wmma_kernel<<<dim3(3 * DD / 64, BQ / 128), blk, 0, stream>>>(
        X, l_qkv_w, l_qkv_b, QKV, BQ, 3 * DD, DD, 0);
    attn_kernel<<<dim3(QQ / 16, NHH, BB), blk, 0, stream>>>(QKV, ATTO);
    gemm_wmma_kernel<<<dim3(DD / 64, BQ / 128), blk, 0, stream>>>(
        ATTO, l_ao_w, l_ao_b, T2, BQ, DD, DD, 0);
    ln_residual_kernel<<<dim3(BQ / 8), blk, 0, stream>>>(
        X, T2, n2_g + (size_t)lid * DD, n2_b + (size_t)lid * DD, BQ);

    // ---- deformable cross-attention
    gemm_wmma_kernel<<<dim3(DD / 64, BS / 128), blk, 0, stream>>>(
        src, l_val_w, l_val_b, VAL, BS, DD, DD, 0);           // value projection
    gemm_wmma_kernel<<<dim3(256 / 64, BQ / 128), blk, 0, stream>>>(
        X, l_off_w, l_off_b, OFFR, BQ, 256, DD, 0);           // offsets
    gemm_wmma_kernel<<<dim3(128 / 64, BQ / 128), blk, 0, stream>>>(
        X, l_aw_w, l_aw_b, AWR, BQ, 128, DD, 0);              // attn weights
    prep_sampling_kernel<<<dim3((BQ * NHH + 255) / 256), blk, 0, stream>>>(
        OFFR, AWR, REF, valid, LOC, AWN);
    deform_sample_kernel<<<dim3(BQ), blk, 0, stream>>>(VAL, LOC, AWN, DEF);
    gemm_wmma_kernel<<<dim3(DD / 64, BQ / 128), blk, 0, stream>>>(
        DEF, l_out_w, l_out_b, T2, BQ, DD, DD, 0);
    ln_residual_kernel<<<dim3(BQ / 8), blk, 0, stream>>>(
        X, T2, n1_g + (size_t)lid * DD, n1_b + (size_t)lid * DD, BQ);

    // ---- FFN + add&LN(n3)
    gemm_wmma_kernel<<<dim3(DFFF / 64, BQ / 128), blk, 0, stream>>>(
        X, l_f1_w, l_f1_b, HID, BQ, DFFF, DD, 1);
    gemm_wmma_kernel<<<dim3(DD / 64, BQ / 128), blk, 0, stream>>>(
        HID, l_f2_w, l_f2_b, T2, BQ, DD, DFFF, 0);
    ln_residual_kernel<<<dim3(BQ / 8), blk, 0, stream>>>(
        X, T2, n3_g + (size_t)lid * DD, n3_b + (size_t)lid * DD, BQ);

    // ---- iterative reference-point refinement
    coord_update_kernel<<<dim3((BQ + 255) / 256), blk, 0, stream>>>(
        X, l_cw, l_cb, REF);
  }

  finalize_kernel<<<dim3(BQ), blk, 0, stream>>>(X, REF, cls_w, cls_b, (float*)d_out);
}
